// IGMC_21620865368721
// MI455X (gfx1250) — matrix operand order repaired and checked by reading, not compiled, not run on hardware
//
#include <hip/hip_runtime.h>
#include <math.h>

typedef __attribute__((ext_vector_type(16))) _Float16     v16h;
typedef __attribute__((ext_vector_type(8)))  float        v8f;
typedef __attribute__((ext_vector_type(8)))  unsigned int v8u;

#define NNODES 100000
#define NEDGES 2000000
#define NPAIRS 20000
#define NREL   5
#define NBASES 4
#define HID    32
#define WCOLS  (NREL*HID + HID)   /* 192: [r*H+h | root block] */
#define NTL    (WCOLS/16)         /* 12 column tiles, layer GEMM */
#define HCATW  128

// Assemble a 16-half WMMA fragment from two b128 loads (8 dwords = 16 f16).
__device__ inline v16h igmc_pack16h(uint4 lo, uint4 hi) {
    v8u u;
    u[0] = lo.x; u[1] = lo.y; u[2] = lo.z; u[3] = lo.w;
    u[4] = hi.x; u[5] = hi.y; u[6] = hi.z; u[7] = hi.w;
    return __builtin_bit_cast(v16h, u);
}

// ---------------------------------------------------------------------------
// Build f16 input matrix for layer 0: [N,32] = x[:, :4] padded with zeros.
__global__ void igmc_build_x0(const float* __restrict__ x, _Float16* __restrict__ X) {
    int tid = blockIdx.x * blockDim.x + threadIdx.x;
    if (tid >= NNODES * HID) return;
    int n = tid >> 5, c = tid & 31;
    X[tid] = (_Float16)((c < 4) ? x[n * 4 + c] : 0.0f);
}

// Dense weight value W[k, j] for one layer:
//   cols [0,160): W_r = sum_b comp[r,b]*basis[b]   (rows >= in_dim are zero)
//   cols [160,192): root
__device__ inline float igmc_wval(const float* __restrict__ basis,
                                  const float* __restrict__ comp,
                                  const float* __restrict__ root,
                                  int in_dim, int k, int j) {
    if (k >= in_dim) return 0.0f;
    if (j < NREL * HID) {
        int r = j / HID, h = j % HID;
        float v = 0.0f;
#pragma unroll
        for (int b = 0; b < NBASES; ++b)
            v += comp[r * NBASES + b] * basis[(b * in_dim + k) * HID + h];
        return v;
    }
    return root[k * HID + (j - NREL * HID)];
}

// Emit the layer weights directly in the per-lane WMMA B-fragment layout:
//   Bsw[((tn*32 + lane)*8 + v)] = pack(W[kb,col], W[kb+1,col])
//   col = tn*16 + (lane&15), kb = 2v + (v&4 ? 8 : 0) + (lane&16 ? 8 : 0)
__global__ void igmc_build_weights_sw(const float* __restrict__ basis,
                                      const float* __restrict__ comp,
                                      const float* __restrict__ root,
                                      int in_dim, unsigned int* __restrict__ Bsw) {
    int tid = blockIdx.x * blockDim.x + threadIdx.x;
    if (tid >= NTL * 32 * 8) return;                  // 3072 dwords
    int v    = tid & 7;
    int lane = (tid >> 3) & 31;
    int tn   = tid >> 8;
    int col  = tn * 16 + (lane & 15);
    int kb   = 2 * v + ((v & 4) ? 8 : 0) + ((lane & 16) ? 8 : 0);
    union { _Float16 h[2]; unsigned int u; } p;
    p.h[0] = (_Float16)igmc_wval(basis, comp, root, in_dim, kb,     col);
    p.h[1] = (_Float16)igmc_wval(basis, comp, root, in_dim, kb + 1, col);
    Bsw[tid] = p.u;
}

// w1[256,128] -> swizzled B fragments, indexed [((kc*8+tn)*32+lane)*8 + v].
__global__ void igmc_cvt_w1_sw(const float* __restrict__ w1,
                               unsigned int* __restrict__ Wsw) {
    int tid = blockIdx.x * blockDim.x + threadIdx.x;
    if (tid >= 8 * 8 * 32 * 8) return;                // 16384 dwords
    int v    = tid & 7;
    int lane = (tid >> 3) & 31;
    int tn   = (tid >> 8) & 7;
    int kc   = tid >> 11;
    int col  = tn * 16 + (lane & 15);
    int kb   = kc * 32 + 2 * v + ((v & 4) ? 8 : 0) + ((lane & 16) ? 8 : 0);
    union { _Float16 h[2]; unsigned int u; } p;
    p.h[0] = (_Float16)w1[kb * 128 + col];
    p.h[1] = (_Float16)w1[(kb + 1) * 128 + col];
    Wsw[tid] = p.u;
}

// ---------------------------------------------------------------------------
// WMMA GEMM: Y[N,192] = X[N,32](f16) @ W[32,192](f16, pre-swizzled), f32 out.
// One wave per 16-row strip: A fragment loaded once (2x b128), reused for all
// 12 column tiles -> 12 v_wmma per wave.
__global__ void igmc_gemm_layer_wmma(const _Float16* __restrict__ X,
                                     const unsigned int* __restrict__ Bsw,
                                     float* __restrict__ Y) {
    const int lane = threadIdx.x & 31;
    const int wave = (blockIdx.x * blockDim.x + threadIdx.x) >> 5;
    if (wave >= NNODES / 16) return;
    const int row = wave * 16 + (lane & 15);
    const int sel = (lane >> 4) & 1;           // lane-half K interleave select
    const uint4* Xp = (const uint4*)X;         // 4 uint4 per 32-half row
    uint4 a0 = Xp[row * 4 + sel];
    uint4 a1 = Xp[row * 4 + sel + 2];
    v16h a = igmc_pack16h(a0, a1);
    const uint4* Bp = (const uint4*)Bsw;
    const int mbase = wave * 16 + ((lane & 16) ? 8 : 0);
    const int nlo = lane & 15;
#pragma unroll
    for (int tn = 0; tn < NTL; ++tn) {
        int bi = (tn * 32 + lane) * 2;
        v16h b = igmc_pack16h(Bp[bi], Bp[bi + 1]);
        v8f c = {};
        c = __builtin_amdgcn_wmma_f32_16x16x32_f16(false, a, false, b, (short)0, c, false, false);
        int col = tn * 16 + nlo;
#pragma unroll
        for (int r = 0; r < 8; ++r)
            Y[(size_t)(mbase + r) * WCOLS + col] = c[r];
    }
}

// ---------------------------------------------------------------------------
// Per-(node,relation) in-degree counts (same every layer -> once per call).
__global__ void igmc_edge_count(const int* __restrict__ dst,
                                const int* __restrict__ et,
                                float* __restrict__ cnt) {
    int e = blockIdx.x * blockDim.x + threadIdx.x;
    if (e >= NEDGES) return;
    atomicAdd(&cnt[dst[e] * NREL + et[e]], 1.0f);
}

// Gather xr[src,etype,:] and scatter-add into sums[(dst*R+etype),:].
// Lane h of each edge-wave handles one of the 32 hidden channels.
__global__ void igmc_edge_scatter(const int* __restrict__ src,
                                  const int* __restrict__ dst,
                                  const int* __restrict__ et,
                                  const float* __restrict__ xrall,
                                  float* __restrict__ sums) {
    long tid = (long)blockIdx.x * blockDim.x + threadIdx.x;
    if (tid >= (long)NEDGES * HID) return;
    int e = (int)(tid >> 5), h = (int)(tid & 31);
    int s = src[e], d = dst[e], r = et[e];
    float v = xrall[(size_t)s * WCOLS + r * HID + h];
    atomicAdd(&sums[(size_t)(d * NREL + r) * HID + h], v);
}

// mean per (node,rel), sum over rel, + root + bias, tanh; write f32 into hcat
// column block and f16 into the next layer's input.
__global__ void igmc_finalize_layer(const float* __restrict__ sums,
                                    const float* __restrict__ cnt,
                                    const float* __restrict__ xrall,
                                    const float* __restrict__ bias,
                                    float* __restrict__ hcat,
                                    _Float16* __restrict__ Xnext,
                                    int layer) {
    int tid = blockIdx.x * blockDim.x + threadIdx.x;
    if (tid >= NNODES * HID) return;
    int n = tid >> 5, h = tid & 31;
    float acc = 0.0f;
#pragma unroll
    for (int r = 0; r < NREL; ++r) {
        float cn = cnt[n * NREL + r];
        acc += sums[(size_t)(n * NREL + r) * HID + h] / fmaxf(cn, 1.0f);
    }
    acc += xrall[(size_t)n * WCOLS + NREL * HID + h] + bias[h];
    float o = tanhf(acc);
    hcat[(size_t)n * HCATW + layer * HID + h] = o;
    Xnext[(size_t)n * HID + h] = (_Float16)o;
}

// ---------------------------------------------------------------------------
// g[G,256] = [hcat[user] | hcat[item]] in f16
__global__ void igmc_build_g(const float* __restrict__ hcat,
                             const int* __restrict__ uidx,
                             const int* __restrict__ iidx,
                             _Float16* __restrict__ Gm) {
    int tid = blockIdx.x * blockDim.x + threadIdx.x;
    if (tid >= NPAIRS * 256) return;
    int g = tid >> 8, j = tid & 255;
    int rowN = (j < 128) ? uidx[g] : iidx[g];
    Gm[tid] = (_Float16)hcat[(size_t)rowN * HCATW + (j & 127)];
}

// WMMA GEMM: O1[G,128] = relu(g[G,256] @ w1[256,128] + b1).
// One wave per 16-row strip; K outer (8 chunks, A loaded once per chunk),
// all 8 column tiles kept in accumulators -> 64 v_wmma per wave.
__global__ void igmc_gemm_mlp_wmma(const _Float16* __restrict__ Gm,
                                   const unsigned int* __restrict__ Wsw,
                                   const float* __restrict__ b1,
                                   float* __restrict__ O1) {
    const int lane = threadIdx.x & 31;
    const int wave = (blockIdx.x * blockDim.x + threadIdx.x) >> 5;
    if (wave >= NPAIRS / 16) return;
    const int row = wave * 16 + (lane & 15);
    const int sel = (lane >> 4) & 1;
    const uint4* Gp = (const uint4*)Gm;        // 32 uint4 per 256-half row
    const uint4* Bp = (const uint4*)Wsw;
    v8f c[8] = {};
#pragma unroll
    for (int kc = 0; kc < 8; ++kc) {
        uint4 a0 = Gp[row * 32 + kc * 4 + sel];
        uint4 a1 = Gp[row * 32 + kc * 4 + sel + 2];
        v16h a = igmc_pack16h(a0, a1);
#pragma unroll
        for (int tn = 0; tn < 8; ++tn) {
            int bi = ((kc * 8 + tn) * 32 + lane) * 2;
            v16h b = igmc_pack16h(Bp[bi], Bp[bi + 1]);
            c[tn] = __builtin_amdgcn_wmma_f32_16x16x32_f16(false, a, false, b, (short)0, c[tn], false, false);
        }
    }
    const int mbase = wave * 16 + ((lane & 16) ? 8 : 0);
    const int nlo = lane & 15;
#pragma unroll
    for (int tn = 0; tn < 8; ++tn) {
        int col = tn * 16 + nlo;
        float bias = b1[col];
#pragma unroll
        for (int r = 0; r < 8; ++r) {
            float vv = c[tn][r] + bias;
            O1[(size_t)(mbase + r) * 128 + col] = vv > 0.0f ? vv : 0.0f;
        }
    }
}

// out[g] = O1[g,:] . w2 + b2
__global__ void igmc_final_dot(const float* __restrict__ O1,
                               const float* __restrict__ w2,
                               const float* __restrict__ b2,
                               float* __restrict__ out) {
    int g = blockIdx.x * blockDim.x + threadIdx.x;
    if (g >= NPAIRS) return;
    float acc = b2[0];
#pragma unroll 4
    for (int j = 0; j < 128; ++j)
        acc += O1[(size_t)g * 128 + j] * w2[j];
    out[g] = acc;
}

// ---------------------------------------------------------------------------
extern "C" void kernel_launch(void* const* d_in, const int* in_sizes, int n_in,
                              void* d_out, int out_size, void* d_ws, size_t ws_size,
                              hipStream_t stream) {
    (void)in_sizes; (void)n_in; (void)out_size; (void)ws_size;
    const float* x       = (const float*)d_in[0];
    const int*   eidx    = (const int*)  d_in[1];
    const int*   etype   = (const int*)  d_in[2];
    const int*   uidx    = (const int*)  d_in[3];
    const int*   iidx    = (const int*)  d_in[4];
    const float* basis0  = (const float*)d_in[5];
    const float* comp0   = (const float*)d_in[6];
    const float* root0   = (const float*)d_in[7];
    const float* bias0   = (const float*)d_in[8];
    const float* basis_h = (const float*)d_in[9];
    const float* comp_h  = (const float*)d_in[10];
    const float* root_h  = (const float*)d_in[11];
    const float* bias_h  = (const float*)d_in[12];
    const float* w1      = (const float*)d_in[13];
    const float* b1      = (const float*)d_in[14];
    const float* w2      = (const float*)d_in[15];
    const float* b2      = (const float*)d_in[16];
    const int* src = eidx;
    const int* dst = eidx + NEDGES;
    float* out = (float*)d_out;

    // -------- workspace carve-out (~221 MB) --------
    char* ws = (char*)d_ws;
    size_t off = 0;
    auto take = [&](size_t bytes) -> char* {
        char* p = ws + off;
        off = (off + bytes + 255) & ~(size_t)255;
        return p;
    };
    float*        sums  = (float*)       take((size_t)NNODES * NREL * HID * 4); // 64 MB
    float*        cnt   = (float*)       take((size_t)NNODES * NREL * 4);       //  2 MB
    _Float16*     Xf16  = (_Float16*)    take((size_t)NNODES * HID * 2);        //6.4 MB
    unsigned int* Bsw   = (unsigned int*)take((size_t)NTL * 32 * 8 * 4);        // 12 KB
    float*        xrall = (float*)       take((size_t)NNODES * WCOLS * 4);      // 77 MB
    float*        hcat  = (float*)       take((size_t)NNODES * HCATW * 4);      // 51 MB
    unsigned int* Wsw   = (unsigned int*)take((size_t)8 * 8 * 32 * 8 * 4);      // 64 KB
    _Float16*     gf16  = (_Float16*)    take((size_t)NPAIRS * 256 * 2);        // 10 MB
    float*        out1  = (float*)       take((size_t)NPAIRS * 128 * 4);        // 10 MB

    const int TB = 256;
    // degree counts (constant across layers)
    hipMemsetAsync(cnt, 0, (size_t)NNODES * NREL * 4, stream);
    igmc_edge_count<<<(NEDGES + TB - 1) / TB, TB, 0, stream>>>(dst, etype, cnt);
    // layer-0 input + MLP weights (pre-swizzled f16 B fragments)
    igmc_build_x0<<<(NNODES * HID) / TB, TB, 0, stream>>>(x, Xf16);
    igmc_cvt_w1_sw<<<(8 * 8 * 32 * 8) / TB, TB, 0, stream>>>(w1, Wsw);

    const int layer_waves  = NNODES / 16;                       // 6250
    const int layer_blocks = (layer_waves + 3) / 4;             // 4 waves/block of 128
    for (int l = 0; l < 4; ++l) {
        const float* bas  = (l == 0) ? basis0 : basis_h + (size_t)(l - 1) * NBASES * HID * HID;
        const float* cmp  = (l == 0) ? comp0  : comp_h  + (size_t)(l - 1) * NREL * NBASES;
        const float* rt   = (l == 0) ? root0  : root_h  + (size_t)(l - 1) * HID * HID;
        const float* bia  = (l == 0) ? bias0  : bias_h  + (size_t)(l - 1) * HID;
        const int in_dim  = (l == 0) ? 4 : HID;

        igmc_build_weights_sw<<<(NTL * 32 * 8 + TB - 1) / TB, TB, 0, stream>>>(bas, cmp, rt, in_dim, Bsw);
        igmc_gemm_layer_wmma<<<layer_blocks, 128, 0, stream>>>(Xf16, Bsw, xrall);
        hipMemsetAsync(sums, 0, (size_t)NNODES * NREL * HID * 4, stream);
        igmc_edge_scatter<<<(int)(((long)NEDGES * HID) / TB), TB, 0, stream>>>(src, dst, etype, xrall, sums);
        igmc_finalize_layer<<<(NNODES * HID) / TB, TB, 0, stream>>>(sums, cnt, xrall, bia, hcat, Xf16, l);
    }

    igmc_build_g<<<(NPAIRS * 256) / TB, TB, 0, stream>>>(hcat, uidx, iidx, gf16);
    const int mlp_waves = NPAIRS / 16;                          // 1250
    igmc_gemm_mlp_wmma<<<(mlp_waves + 3) / 4, 128, 0, stream>>>(gf16, Wsw, b1, out1);
    igmc_final_dot<<<(NPAIRS + TB - 1) / TB, TB, 0, stream>>>(out1, w2, b2, out);
}
